// ns_informer_model_11218454577563
// MI455X (gfx1250) — matrix operand order repaired and checked by reading, not compiled, not run on hardware
//
#include <hip/hip_runtime.h>
#include <math.h>

// ---------------- model dimensions ----------------
#define B_     32
#define S_     1536
#define LBL_   768
#define PRD_   768
#define C_     32
#define DM_    512
#define H_     8
#define HD_    64
#define DFF_   2048
#define TF_    4
#define EL_    2
#define PH_    128
#define DEC_   1536
#define U_     40     // FACTOR * ceil(ln(1536)) = 5*8
#define UPART_ 40

// ---------------- WMMA GEMM tiling ----------------
#define BM 128
#define BN 128
#define KC 32
#define LSTR 40   // padded LDS row stride (fp16 elems): 80B rows -> 16B-aligned frags

typedef _Float16 v16h __attribute__((ext_vector_type(16)));
typedef _Float16 v8h  __attribute__((ext_vector_type(8)));
typedef float    v8f  __attribute__((ext_vector_type(8)));

// =====================================================================
// WMMA GEMM:  C[M,N] = act( A[M,K] * W[N,K]^T + bias[N] )
// K must be a multiple of 32 (holds for 512 / 2048 / 96).
// act: 0=none, 1=relu, 2=gelu(exact)
// =====================================================================
__global__ void __launch_bounds__(256)
gemm_wmma_kernel(const float* __restrict__ A, const float* __restrict__ W,
                 const float* __restrict__ bias, float* __restrict__ Cm,
                 int M, int N, int K, int act)
{
    __shared__ _Float16 As[BM * LSTR];
    __shared__ _Float16 Ws[BN * LSTR];

    const int tid  = threadIdx.x;
    const int lane = tid & 31;
    const int wave = tid >> 5;      // 8 waves
    const int wm   = wave & 3;      // 4 waves along M (32 rows each)
    const int wn   = wave >> 2;     // 2 waves along N (64 cols each)
    const int m0   = blockIdx.y * BM;
    const int n0   = blockIdx.x * BN;

    const int half = lane >> 4;     // 0/1
    const int l15  = lane & 15;

    v8f acc[2][4] = {};

    for (int k0 = 0; k0 < K; k0 += KC) {
        // stage A tile (BM x 32) as f16 into LDS
        for (int c = tid; c < BM * (KC / 4); c += 256) {
            int r  = c >> 3;
            int kk = (c & 7) << 2;
            float4 v = {0.f, 0.f, 0.f, 0.f};
            int gm = m0 + r;
            if (gm < M) v = *(const float4*)(A + (size_t)gm * K + k0 + kk);
            _Float16* d = &As[r * LSTR + kk];
            d[0] = (_Float16)v.x; d[1] = (_Float16)v.y;
            d[2] = (_Float16)v.z; d[3] = (_Float16)v.w;
        }
        // stage W tile (BN x 32)
        for (int c = tid; c < BN * (KC / 4); c += 256) {
            int r  = c >> 3;
            int kk = (c & 7) << 2;
            float4 v = {0.f, 0.f, 0.f, 0.f};
            int gn = n0 + r;
            if (gn < N) v = *(const float4*)(W + (size_t)gn * K + k0 + kk);
            _Float16* d = &Ws[r * LSTR + kk];
            d[0] = (_Float16)v.x; d[1] = (_Float16)v.y;
            d[2] = (_Float16)v.z; d[3] = (_Float16)v.w;
        }
        __syncthreads();

        // A fragments: lane l holds row (l&15), K chunks {half*8..+8, 16+half*8..+8}
        v16h afrag[2];
        #pragma unroll
        for (int i = 0; i < 2; ++i) {
            const _Float16* bp = &As[(wm * 32 + i * 16 + l15) * LSTR + half * 8];
            v8h lo = *(const v8h*)(bp);
            v8h hi = *(const v8h*)(bp + 16);
            afrag[i] = __builtin_shufflevector(lo, hi, 0,1,2,3,4,5,6,7,8,9,10,11,12,13,14,15);
        }
        // B fragments: lane holds column (l&15), K chunks {half*8..+8, 16+half*8..+8}
        v16h bfrag[4];
        #pragma unroll
        for (int j = 0; j < 4; ++j) {
            const _Float16* bp = &Ws[(wn * 64 + j * 16 + l15) * LSTR + half * 8];
            v8h lo = *(const v8h*)(bp);
            v8h hi = *(const v8h*)(bp + 16);
            bfrag[j] = __builtin_shufflevector(lo, hi, 0,1,2,3,4,5,6,7,8,9,10,11,12,13,14,15);
        }

        #pragma unroll
        for (int i = 0; i < 2; ++i)
            #pragma unroll
            for (int j = 0; j < 4; ++j)
                acc[i][j] = __builtin_amdgcn_wmma_f32_16x16x32_f16(
                    false, afrag[i], false, bfrag[j], (short)0, acc[i][j], false, false);
        __syncthreads();
    }

    // epilogue per C/D layout: VGPR r -> M = (lane>>4)*8 + r ; N = lane&15
    const int mh = half * 8;
    #pragma unroll
    for (int i = 0; i < 2; ++i) {
        #pragma unroll
        for (int j = 0; j < 4; ++j) {
            int ncol = n0 + wn * 64 + j * 16 + l15;
            if (ncol >= N) continue;
            float bv = bias ? bias[ncol] : 0.f;
            #pragma unroll
            for (int r = 0; r < 8; ++r) {
                int mrow = m0 + wm * 32 + i * 16 + mh + r;
                if (mrow >= M) continue;
                float v = acc[i][j][r] + bv;
                if (act == 1)      v = fmaxf(v, 0.f);
                else if (act == 2) v = 0.5f * v * (1.f + erff(v * 0.70710678f));
                Cm[(size_t)mrow * N + ncol] = v;
            }
        }
    }
}

// =====================================================================
// elementwise / reduction kernels
// =====================================================================
__global__ void __launch_bounds__(256)
stats_kernel(const float* __restrict__ x, float* __restrict__ mean, float* __restrict__ stdv)
{
    __shared__ float red[256];
    int b = blockIdx.x / C_, c = blockIdx.x % C_;
    const float* p = x + (size_t)b * S_ * C_ + c;
    int tid = threadIdx.x;
    float s = 0.f;
    for (int l = tid; l < S_; l += 256) s += p[(size_t)l * C_];
    red[tid] = s; __syncthreads();
    for (int o = 128; o; o >>= 1) { if (tid < o) red[tid] += red[tid + o]; __syncthreads(); }
    float mu = red[0] / (float)S_;
    __syncthreads();
    float v = 0.f;
    for (int l = tid; l < S_; l += 256) { float d = p[(size_t)l * C_] - mu; v += d * d; }
    red[tid] = v; __syncthreads();
    for (int o = 128; o; o >>= 1) { if (tid < o) red[tid] += red[tid + o]; __syncthreads(); }
    if (tid == 0) { mean[blockIdx.x] = mu; stdv[blockIdx.x] = sqrtf(red[0] / (float)S_ + 1e-5f); }
}

__global__ void normalize_kernel(const float* __restrict__ x, const float* __restrict__ mean,
                                 const float* __restrict__ stdv, float* __restrict__ xe)
{
    size_t idx = (size_t)blockIdx.x * 256 + threadIdx.x;
    if (idx >= (size_t)B_ * S_ * C_) return;
    int c = idx % C_;
    int b = idx / ((size_t)S_ * C_);
    xe[idx] = (x[idx] - mean[b * C_ + c]) / stdv[b * C_ + c];
}

__global__ void xdecnew_kernel(const float* __restrict__ xe, float* __restrict__ xd)
{
    size_t idx = (size_t)blockIdx.x * 256 + threadIdx.x;
    if (idx >= (size_t)B_ * DEC_ * C_) return;
    int c = idx % C_;
    int l = (idx / C_) % DEC_;
    int b = idx / ((size_t)C_ * DEC_);
    float v = 0.f;
    if (l < LBL_) v = xe[((size_t)b * S_ + (S_ - LBL_) + l) * C_ + c];
    xd[idx] = v;
}

// projector conv over channel dim (circular), weight (S,3)
__global__ void pconv_kernel(const float* __restrict__ x, const float* __restrict__ w,
                             float* __restrict__ out)
{
    int idx = blockIdx.x * 256 + threadIdx.x;
    if (idx >= B_ * C_) return;
    int e = idx % C_, b = idx / C_;
    int em = (e == 0) ? C_ - 1 : e - 1;
    int ep = (e == C_ - 1) ? 0 : e + 1;
    const float* xb = x + (size_t)b * S_ * C_;
    float s = 0.f;
    for (int t = 0; t < S_; ++t) {
        const float* xr = xb + (size_t)t * C_;
        const float* wr = w + t * 3;
        s += xr[em] * wr[0] + xr[e] * wr[1] + xr[ep] * wr[2];
    }
    out[idx] = s;
}

__global__ void hcat_kernel(const float* __restrict__ a, const float* __restrict__ st,
                            float* __restrict__ h)
{
    int idx = blockIdx.x * 256 + threadIdx.x;
    if (idx >= B_ * 2 * C_) return;
    int i = idx % (2 * C_), b = idx / (2 * C_);
    h[idx] = (i < C_) ? a[b * C_ + i] : st[b * C_ + (i - C_)];
}

// tiny GEMM for projector heads. act: 0 none, 1 relu, 3 exp
__global__ void small_gemm_kernel(const float* __restrict__ A, const float* __restrict__ W,
                                  const float* __restrict__ bias, float* __restrict__ Cm,
                                  int M, int N, int K, int act)
{
    int idx = blockIdx.x * 256 + threadIdx.x;
    if (idx >= M * N) return;
    int m = idx / N, n = idx % N;
    float s = bias ? bias[n] : 0.f;
    for (int k = 0; k < K; ++k) s += A[(size_t)m * K + k] * W[(size_t)n * K + k];
    if (act == 1)      s = fmaxf(s, 0.f);
    else if (act == 3) s = __expf(s);
    Cm[idx] = s;
}

// circular-in-time im2col: col[(b,l), c*3+j] = x[b, wrap(l+j-1), c]
__global__ void im2col_kernel(const float* __restrict__ x, float* __restrict__ col, int L)
{
    int idx = blockIdx.x * 256 + threadIdx.x;
    if (idx >= B_ * L * C_) return;
    int c = idx % C_;
    int l = (idx / C_) % L;
    int b = idx / (C_ * L);
    int lm = (l == 0) ? L - 1 : l - 1;
    int lp = (l == L - 1) ? 0 : l + 1;
    const float* xb = x + (size_t)b * L * C_;
    float* crow = col + ((size_t)b * L + l) * (3 * C_) + c * 3;
    crow[0] = xb[(size_t)lm * C_ + c];
    crow[1] = xb[(size_t)l  * C_ + c];
    crow[2] = xb[(size_t)lp * C_ + c];
}

// x += pos_emb(l,d) + mark(b,l,:) @ Wt[d,:]
__global__ void embed_finish_kernel(float* __restrict__ x, const float* __restrict__ mark,
                                    const float* __restrict__ wt, int L)
{
    size_t idx = (size_t)blockIdx.x * 256 + threadIdx.x;
    if (idx >= (size_t)B_ * L * DM_) return;
    int d = idx % DM_;
    size_t bl = idx / DM_;
    int l = bl % L;
    float div = __expf(-(9.210340372f / (float)DM_) * (float)(d & ~1));
    float arg = (float)l * div;
    float pe = (d & 1) ? __cosf(arg) : __sinf(arg);
    const float* mk = mark + bl * TF_;
    float t = 0.f;
    #pragma unroll
    for (int j = 0; j < TF_; ++j) t += mk[j] * wt[d * TF_ + j];
    x[idx] += pe + t;
}

// y = LN(x + r) * w + b   (r may be null; y may alias x)
__global__ void __launch_bounds__(256)
ln_res_kernel(const float* __restrict__ x, const float* __restrict__ r,
              const float* __restrict__ w, const float* __restrict__ bsh,
              float* __restrict__ y, int D)
{
    __shared__ float red[256];
    size_t row = blockIdx.x;
    int tid = threadIdx.x;
    const float* xp = x + row * (size_t)D;
    const float* rp = r ? r + row * (size_t)D : nullptr;
    float* yp = y + row * (size_t)D;
    float s = 0.f;
    for (int d = tid; d < D; d += 256) { float v = xp[d] + (rp ? rp[d] : 0.f); yp[d] = v; s += v; }
    red[tid] = s; __syncthreads();
    for (int o = 128; o; o >>= 1) { if (tid < o) red[tid] += red[tid + o]; __syncthreads(); }
    float mu = red[0] / (float)D;
    __syncthreads();
    float vs = 0.f;
    for (int d = tid; d < D; d += 256) { float dd = yp[d] - mu; vs += dd * dd; }
    red[tid] = vs; __syncthreads();
    for (int o = 128; o; o >>= 1) { if (tid < o) red[tid] += red[tid + o]; __syncthreads(); }
    float inv = rsqrtf(red[0] / (float)D + 1e-5f);
    for (int d = tid; d < D; d += 256) yp[d] = (yp[d] - mu) * inv * w[d] + bsh[d];
}

// ---------------- ProbSparse attention (tail kernels, ~2% of FLOPs) ---------
__device__ __forceinline__ unsigned hash_u32(unsigned x)
{
    x ^= x >> 16; x *= 0x7feb352dU; x ^= x >> 15; x *= 0x846ca68bU; x ^= x >> 16;
    return x;
}

// sparsity measure M[b,h,l] over UPART sampled keys
__global__ void prob_m_kernel(const float* __restrict__ q, const float* __restrict__ k,
                              float* __restrict__ Mout, int seed, int Lq, int Lk)
{
    int idx = blockIdx.x * 256 + threadIdx.x;
    if (idx >= B_ * H_ * Lq) return;
    int l = idx % Lq; int bh = idx / Lq; int h = bh % H_; int b = bh / H_;
    const float* qr = q + ((size_t)b * Lq + l) * DM_ + h * HD_;
    float mx = -3.4e38f, sm = 0.f;
    for (int s = 0; s < UPART_; ++s) {
        unsigned ki = hash_u32((unsigned)seed * 2654435761u ^ (unsigned)(l * 40503 + s)) % (unsigned)Lk;
        const float* kr = k + ((size_t)b * Lk + ki) * DM_ + h * HD_;
        float d = 0.f;
        for (int t = 0; t < HD_; ++t) d += qr[t] * kr[t];
        mx = fmaxf(mx, d); sm += d;
    }
    Mout[idx] = mx - sm / (float)Lk;
}

// iterative top-U argmax per (b,h); destroys Mbuf
__global__ void __launch_bounds__(256)
topk_kernel(float* __restrict__ Mbuf, int* __restrict__ top, int Lq)
{
    __shared__ float rv[256];
    __shared__ int   ri[256];
    int bh = blockIdx.x;
    int tid = threadIdx.x;
    float* Mp = Mbuf + (size_t)bh * Lq;
    for (int it = 0; it < U_; ++it) {
        float best = -3.4e38f; int bi = 0;
        for (int l = tid; l < Lq; l += 256) {
            float v = Mp[l];
            if (v > best) { best = v; bi = l; }
        }
        rv[tid] = best; ri[tid] = bi; __syncthreads();
        for (int o = 128; o; o >>= 1) {
            if (tid < o && rv[tid + o] > rv[tid]) { rv[tid] = rv[tid + o]; ri[tid] = ri[tid + o]; }
            __syncthreads();
        }
        if (tid == 0) { top[bh * U_ + it] = ri[0]; Mp[ri[0]] = -3.4e38f; }
        __syncthreads();
    }
}

// default context: mean(v) broadcast (enc/cross) or cumsum(v) (masked self)
__global__ void ctx_init_kernel(const float* __restrict__ v, float* __restrict__ o,
                                int Lk, int Lq, int masked)
{
    int idx = blockIdx.x * 256 + threadIdx.x;
    if (idx >= B_ * H_ * HD_) return;
    int d = idx % HD_; int bh = idx / HD_; int h = bh % H_; int b = bh / H_;
    const float* vp = v + (size_t)b * Lk * DM_ + h * HD_ + d;
    float* op = o + (size_t)b * Lq * DM_ + h * HD_ + d;
    if (masked) {
        float run = 0.f;
        for (int l = 0; l < Lq; ++l) { run += vp[(size_t)l * DM_]; op[(size_t)l * DM_] = run; }
    } else {
        float s = 0.f;
        for (int l = 0; l < Lk; ++l) s += vp[(size_t)l * DM_];
        s /= (float)Lk;
        for (int l = 0; l < Lq; ++l) op[(size_t)l * DM_] = s;
    }
}

// scores[b,h,u,k] = (q_top . k) * tau + delta, scaled; causal mask optional
__global__ void __launch_bounds__(256)
scores_kernel(const float* __restrict__ q, const float* __restrict__ k,
              const int* __restrict__ top, const float* __restrict__ tau,
              const float* __restrict__ delta, float* __restrict__ sc,
              int Lq, int Lk, int masked)
{
    __shared__ float qs[HD_];
    int rid = blockIdx.x;
    int u = rid % U_; int bh = rid / U_; int h = bh % H_; int b = bh / H_;
    int ql = top[bh * U_ + u];
    const float* qr = q + ((size_t)b * Lq + ql) * DM_ + h * HD_;
    if (threadIdx.x < HD_) qs[threadIdx.x] = qr[threadIdx.x];
    __syncthreads();
    float tv = tau ? tau[b] : 1.f;
    float* srow = sc + (size_t)rid * Lk;
    for (int kk = threadIdx.x; kk < Lk; kk += 256) {
        float val;
        if (masked && kk > ql) {
            val = -INFINITY;
        } else {
            const float* kr = k + ((size_t)b * Lk + kk) * DM_ + h * HD_;
            float d = 0.f;
            for (int t = 0; t < HD_; ++t) d += qs[t] * kr[t];
            float dv = delta ? delta[(size_t)b * Lk + kk] : 0.f;
            val = (d * tv + dv) * 0.125f;   // 1/sqrt(64)
        }
        srow[kk] = val;
    }
}

__global__ void __launch_bounds__(256)
softmax_kernel(float* __restrict__ sc, int L)
{
    __shared__ float red[256];
    float* row = sc + (size_t)blockIdx.x * L;
    int tid = threadIdx.x;
    float mx = -3.4e38f;
    for (int i = tid; i < L; i += 256) mx = fmaxf(mx, row[i]);
    red[tid] = mx; __syncthreads();
    for (int o = 128; o; o >>= 1) { if (tid < o) red[tid] = fmaxf(red[tid], red[tid + o]); __syncthreads(); }
    mx = red[0];
    __syncthreads();
    float sum = 0.f;
    for (int i = tid; i < L; i += 256) { float e = __expf(row[i] - mx); row[i] = e; sum += e; }
    red[tid] = sum; __syncthreads();
    for (int o = 128; o; o >>= 1) { if (tid < o) red[tid] += red[tid + o]; __syncthreads(); }
    float inv = 1.f / red[0];
    for (int i = tid; i < L; i += 256) row[i] *= inv;
}

__global__ void upd_kernel(const float* __restrict__ attn, const float* __restrict__ v,
                           float* __restrict__ upd, int Lk)
{
    int rid = blockIdx.x;
    int d = threadIdx.x;                       // 64 threads
    int bh = rid / U_; int h = bh % H_; int b = bh / H_;
    const float* arow = attn + (size_t)rid * Lk;
    const float* vp = v + (size_t)b * Lk * DM_ + h * HD_ + d;
    float s = 0.f;
    for (int kk = 0; kk < Lk; ++kk) s += arow[kk] * vp[(size_t)kk * DM_];
    upd[(size_t)rid * HD_ + d] = s;
}

__global__ void scatter_kernel(const float* __restrict__ upd, const int* __restrict__ top,
                               float* __restrict__ o, int Lq)
{
    int rid = blockIdx.x;
    int d = threadIdx.x;
    int u = rid % U_; int bh = rid / U_; int h = bh % H_; int b = bh / H_;
    int l = top[bh * U_ + u];
    o[((size_t)b * Lq + l) * DM_ + h * HD_ + d] = upd[(size_t)rid * HD_ + d];
}

__global__ void final_out_kernel(const float* __restrict__ proj, const float* __restrict__ stdv,
                                 const float* __restrict__ mean, float* __restrict__ out)
{
    int idx = blockIdx.x * 256 + threadIdx.x;
    if (idx >= B_ * PRD_ * C_) return;
    int c = idx % C_; int p = (idx / C_) % PRD_; int b = idx / (C_ * PRD_);
    float v = proj[((size_t)b * DEC_ + LBL_ + p) * C_ + c];
    out[idx] = v * stdv[b * C_ + c] + mean[b * C_ + c];
}

// =====================================================================
// host orchestration
// =====================================================================
struct AttnP { const float *bk, *bo, *bq, *bv, *wk, *wo, *wq, *wv; };
struct EncL  { AttnP attn; const float *b1, *b2, *ln1b, *ln1w, *ln2b, *ln2w, *w1, *w2; };
struct DecL  { const float *b1, *b2; AttnP cross;
               const float *ln1b, *ln1w, *ln2b, *ln2w, *ln3b, *ln3w;
               AttnP self; const float *w1, *w2; };
struct ProjP { const float *b0, *b1, *conv, *w0, *w1, *wout; };

static inline void gemm(hipStream_t st, const float* A, const float* W, const float* bias,
                        float* Cm, int M, int N, int K, int act)
{
    dim3 g((N + BN - 1) / BN, (M + BM - 1) / BM);
    gemm_wmma_kernel<<<g, 256, 0, st>>>(A, W, bias, Cm, M, N, K, act);
}

static inline void run_attn(hipStream_t st, const float* q, const float* k, const float* v,
                            float* o, float* sc, float* Mb, int* top, float* upd,
                            int Lq, int Lk, int masked,
                            const float* tau, const float* delta, int seed)
{
    int nrows = B_ * H_ * Lq;
    prob_m_kernel<<<(nrows + 255) / 256, 256, 0, st>>>(q, k, Mb, seed, Lq, Lk);
    topk_kernel<<<B_ * H_, 256, 0, st>>>(Mb, top, Lq);
    ctx_init_kernel<<<(B_ * H_ * HD_ + 255) / 256, 256, 0, st>>>(v, o, Lk, Lq, masked);
    scores_kernel<<<B_ * H_ * U_, 256, 0, st>>>(q, k, top, tau, delta, sc, Lq, Lk, masked);
    softmax_kernel<<<B_ * H_ * U_, 256, 0, st>>>(sc, Lk);
    upd_kernel<<<B_ * H_ * U_, 64, 0, st>>>(sc, v, upd, Lk);
    scatter_kernel<<<B_ * H_ * U_, 64, 0, st>>>(upd, top, o, Lq);
}

extern "C" void kernel_launch(void* const* d_in, const int* in_sizes, int n_in,
                              void* d_out, int out_size, void* d_ws, size_t ws_size,
                              hipStream_t stream)
{
    (void)in_sizes; (void)n_in; (void)out_size; (void)ws_size;

    const float* x_enc      = (const float*)d_in[0];
    const float* x_mark_enc = (const float*)d_in[1];
    const float* x_dec      = (const float*)d_in[2];  (void)x_dec; // replaced by x_dec_new per ref
    const float* x_mark_dec = (const float*)d_in[3];

    // ---- parameter leaves in JAX pytree (sorted-key) order ----
    int pi = 4;
    auto F = [&]() -> const float* { return (const float*)d_in[pi++]; };
    auto readAttn = [&](AttnP& a) {
        a.bk = F(); a.bo = F(); a.bq = F(); a.bv = F();
        a.wk = F(); a.wo = F(); a.wq = F(); a.wv = F();
    };
    const float* dec_emb_conv = F();
    const float* dec_emb_time = F();
    DecL dl;
    dl.b1 = F(); dl.b2 = F(); readAttn(dl.cross);
    dl.ln1b = F(); dl.ln1w = F(); dl.ln2b = F(); dl.ln2w = F(); dl.ln3b = F(); dl.ln3w = F();
    readAttn(dl.self); dl.w1 = F(); dl.w2 = F();
    const float* dec_normb = F();
    const float* dec_normw = F();
    ProjP deltaP; deltaP.b0 = F(); deltaP.b1 = F(); deltaP.conv = F();
    deltaP.w0 = F(); deltaP.w1 = F(); deltaP.wout = F();
    const float* enc_emb_conv = F();
    const float* enc_emb_time = F();
    EncL el[EL_];
    for (int i = 0; i < EL_; ++i) {
        readAttn(el[i].attn);
        el[i].b1 = F(); el[i].b2 = F();
        el[i].ln1b = F(); el[i].ln1w = F(); el[i].ln2b = F(); el[i].ln2w = F();
        el[i].w1 = F(); el[i].w2 = F();
    }
    const float* enc_normb = F();
    const float* enc_normw = F();
    const float* proj_b = F();
    const float* proj_w = F();
    ProjP tauP; tauP.b0 = F(); tauP.b1 = F(); tauP.conv = F();
    tauP.w0 = F(); tauP.w1 = F(); tauP.wout = F();

    // ---- workspace layout ----
    float* base = (float*)d_ws;
    size_t off = 0;
    auto WS = [&](size_t n) { float* p = base + off; off += (n + 63) & ~(size_t)63; return p; };
    const size_t NX = (size_t)B_ * S_ * DM_;
    float* pX   = WS(NX);
    float* pQ   = WS(NX);
    float* pK   = WS(NX);
    float* pV   = WS(NX);
    float* pO   = WS(NX);
    float* pT   = WS(NX);
    float* pENC = WS(NX);
    float* pBIG = WS((size_t)B_ * S_ * DFF_);       // FFN hidden; overlaid scratch below
    float* pXE  = WS((size_t)B_ * S_ * C_);
    float* pXDN = WS((size_t)B_ * DEC_ * C_);
    float* pMEAN = WS(B_ * C_);
    float* pSTD  = WS(B_ * C_);
    float* pCONV = WS(B_ * C_);
    float* pHCAT = WS(B_ * 2 * C_);
    float* pPH1  = WS(B_ * PH_);
    float* pPH2  = WS(B_ * PH_);
    float* pTAU  = WS(B_);
    float* pDELTA = WS((size_t)B_ * S_);
    float* pPROJ = WS((size_t)B_ * DEC_ * C_);
    // phase-disjoint overlays inside pBIG (attention + im2col never live with FFN hidden)
    float* pSC  = pBIG;                                  // B*H*U*L scores   (~15.7M f)
    float* pM   = pBIG + (size_t)16 * 1024 * 1024;       // B*H*L measure
    float* pUPD = pBIG + (size_t)17 * 1024 * 1024;       // B*H*U*HD
    int*   pTOP = (int*)(pBIG + (size_t)18 * 1024 * 1024);
    float* pI2C = pBIG + (size_t)19 * 1024 * 1024;       // im2col (B*L*96)

    const int BL = B_ * S_;   // == B_*DEC_

    // ---- stats / normalization / decoder seed sequence ----
    stats_kernel<<<B_ * C_, 256, 0, stream>>>(x_enc, pMEAN, pSTD);
    normalize_kernel<<<(int)((NX ? (size_t)B_ * S_ * C_ : 0) + 255) / 256 ? ((size_t)B_ * S_ * C_ + 255) / 256 : 1, 256, 0, stream>>>(x_enc, pMEAN, pSTD, pXE);
    xdecnew_kernel<<<((size_t)B_ * DEC_ * C_ + 255) / 256, 256, 0, stream>>>(pXE, pXDN);

    // ---- tau / delta projectors (on raw x_enc) ----
    pconv_kernel<<<(B_ * C_ + 255) / 256, 256, 0, stream>>>(x_enc, tauP.conv, pCONV);
    hcat_kernel<<<(B_ * 2 * C_ + 255) / 256, 256, 0, stream>>>(pCONV, pSTD, pHCAT);
    small_gemm_kernel<<<(B_ * PH_ + 255) / 256, 256, 0, stream>>>(pHCAT, tauP.w0, tauP.b0, pPH1, B_, PH_, 2 * C_, 1);
    small_gemm_kernel<<<(B_ * PH_ + 255) / 256, 256, 0, stream>>>(pPH1, tauP.w1, tauP.b1, pPH2, B_, PH_, PH_, 1);
    small_gemm_kernel<<<1, 256, 0, stream>>>(pPH2, tauP.wout, nullptr, pTAU, B_, 1, PH_, 3);

    pconv_kernel<<<(B_ * C_ + 255) / 256, 256, 0, stream>>>(x_enc, deltaP.conv, pCONV);
    hcat_kernel<<<(B_ * 2 * C_ + 255) / 256, 256, 0, stream>>>(pCONV, pMEAN, pHCAT);
    small_gemm_kernel<<<(B_ * PH_ + 255) / 256, 256, 0, stream>>>(pHCAT, deltaP.w0, deltaP.b0, pPH1, B_, PH_, 2 * C_, 1);
    small_gemm_kernel<<<(B_ * PH_ + 255) / 256, 256, 0, stream>>>(pPH1, deltaP.w1, deltaP.b1, pPH2, B_, PH_, PH_, 1);
    small_gemm_kernel<<<(B_ * S_ + 255) / 256, 256, 0, stream>>>(pPH2, deltaP.wout, nullptr, pDELTA, B_, S_, PH_, 0);

    // ---- encoder embedding ----
    im2col_kernel<<<(B_ * S_ * C_ + 255) / 256, 256, 0, stream>>>(pXE, pI2C, S_);
    gemm(stream, pI2C, enc_emb_conv, nullptr, pX, BL, DM_, 3 * C_, 0);
    embed_finish_kernel<<<(int)(((size_t)BL * DM_ + 255) / 256), 256, 0, stream>>>(pX, x_mark_enc, enc_emb_time, S_);

    // ---- encoder layers ----
    for (int i = 0; i < EL_; ++i) {
        const EncL& L = el[i];
        gemm(stream, pX, L.attn.wq, L.attn.bq, pQ, BL, DM_, DM_, 0);
        gemm(stream, pX, L.attn.wk, L.attn.bk, pK, BL, DM_, DM_, 0);
        gemm(stream, pX, L.attn.wv, L.attn.bv, pV, BL, DM_, DM_, 0);
        run_attn(stream, pQ, pK, pV, pO, pSC, pM, pTOP, pUPD, S_, S_, 0, pTAU, pDELTA, 100 + i);
        gemm(stream, pO, L.attn.wo, L.attn.bo, pT, BL, DM_, DM_, 0);
        ln_res_kernel<<<BL, 256, 0, stream>>>(pX, pT, L.ln1w, L.ln1b, pO, DM_);
        gemm(stream, pO, L.w1, L.b1, pBIG, BL, DFF_, DM_, 2);
        gemm(stream, pBIG, L.w2, L.b2, pT, BL, DM_, DFF_, 0);
        ln_res_kernel<<<BL, 256, 0, stream>>>(pO, pT, L.ln2w, L.ln2b, pX, DM_);
    }
    ln_res_kernel<<<BL, 256, 0, stream>>>(pX, nullptr, enc_normw, enc_normb, pENC, DM_);

    // ---- decoder embedding ----
    im2col_kernel<<<(B_ * DEC_ * C_ + 255) / 256, 256, 0, stream>>>(pXDN, pI2C, DEC_);
    gemm(stream, pI2C, dec_emb_conv, nullptr, pX, BL, DM_, 3 * C_, 0);
    embed_finish_kernel<<<(int)(((size_t)BL * DM_ + 255) / 256), 256, 0, stream>>>(pX, x_mark_dec, dec_emb_time, DEC_);

    // ---- decoder layer (DL=1) ----
    // masked self-attention
    gemm(stream, pX, dl.self.wq, dl.self.bq, pQ, BL, DM_, DM_, 0);
    gemm(stream, pX, dl.self.wk, dl.self.bk, pK, BL, DM_, DM_, 0);
    gemm(stream, pX, dl.self.wv, dl.self.bv, pV, BL, DM_, DM_, 0);
    run_attn(stream, pQ, pK, pV, pO, pSC, pM, pTOP, pUPD, DEC_, DEC_, 1, pTAU, nullptr, 200);
    gemm(stream, pO, dl.self.wo, dl.self.bo, pT, BL, DM_, DM_, 0);
    ln_res_kernel<<<BL, 256, 0, stream>>>(pX, pT, dl.ln1w, dl.ln1b, pX, DM_);
    // cross-attention against encoder output
    gemm(stream, pX,   dl.cross.wq, dl.cross.bq, pQ, BL, DM_, DM_, 0);
    gemm(stream, pENC, dl.cross.wk, dl.cross.bk, pK, BL, DM_, DM_, 0);
    gemm(stream, pENC, dl.cross.wv, dl.cross.bv, pV, BL, DM_, DM_, 0);
    run_attn(stream, pQ, pK, pV, pO, pSC, pM, pTOP, pUPD, DEC_, S_, 0, pTAU, nullptr, 201);
    gemm(stream, pO, dl.cross.wo, dl.cross.bo, pT, BL, DM_, DM_, 0);
    ln_res_kernel<<<BL, 256, 0, stream>>>(pX, pT, dl.ln2w, dl.ln2b, pX, DM_);
    // FFN
    gemm(stream, pX, dl.w1, dl.b1, pBIG, BL, DFF_, DM_, 2);
    gemm(stream, pBIG, dl.w2, dl.b2, pT, BL, DM_, DFF_, 0);
    ln_res_kernel<<<BL, 256, 0, stream>>>(pX, pT, dl.ln3w, dl.ln3b, pX, DM_);

    // ---- final norm, projection, de-normalization ----
    ln_res_kernel<<<BL, 256, 0, stream>>>(pX, nullptr, dec_normw, dec_normb, pO, DM_);
    gemm(stream, pO, proj_w, proj_b, pPROJ, BL, C_, DM_, 0);
    final_out_kernel<<<(B_ * PRD_ * C_ + 255) / 256, 256, 0, stream>>>(pPROJ, pSTD, pMEAN, (float*)d_out);
}
